// KAN_Rnn_20650202759749
// MI455X (gfx1250) — compile-verified, matrix operand order checked by reading, not run on hardware
//
#include <hip/hip_runtime.h>
#include <cstdint>
#include <cstddef>

// ---------------------------------------------------------------------------
// KAN-RNN for MI455X (gfx1250, wave32, WMMA).
// Each timestep = 2 bf16 WMMA GEMMs (K~4.7k) with fused spline-feature epilogues.
// Weights (~9.5MB bf16) + per-step activations (~19MB bf16) stay in 192MB L2,
// so the loop is matrix-pipe bound -> v_wmma_f32_16x16x32_bf16 everywhere.
// ---------------------------------------------------------------------------

typedef __attribute__((ext_vector_type(16))) __bf16      v16bf;
typedef __attribute__((ext_vector_type(8)))  float       v8f;
typedef __attribute__((ext_vector_type(8)))  unsigned short ushort8;
typedef __attribute__((ext_vector_type(4)))  unsigned int uint4v;

constexpr int BATCH = 1024;
constexpr int SEQ_T = 1024;
constexpr int D_IN  = 12;
constexpr int HID   = 512;
constexpr int NCOEF = 8;                 // G + k spline coefficients
// Layer-1 fused K: [0,524) silu(z), [524,528) zero gap (16B-align bases),
// [528,4720) bases (524*8), [4720,4736) zero pad -> multiple of 32.
constexpr int K1 = 4736;
constexpr int SPL1_OFF = 528;
// Layer-2 fused K: [0,512) silu(h1), [512,4608) bases (512*8). 4608 = 144*32.
constexpr int K2 = 4608;
constexpr int SPL2_OFF = 512;

// ---------------- scalar helpers ----------------

__device__ __forceinline__ unsigned short f2bf(float f) {
  unsigned u = __float_as_uint(f);
  u += 0x7FFFu + ((u >> 16) & 1u);       // round-to-nearest-even
  return (unsigned short)(u >> 16);
}

__device__ __forceinline__ float silu(float x) {
  return x / (1.0f + __expf(-x));
}

// Cubic B-spline bases on uniform grid: knots t[i] = 0.4*i - 2.2, i=0..11.
// Cox-de Boor with compile-time denominators (uniform grid -> p*h).
__device__ __forceinline__ void bases8(float x, float r[NCOEF]) {
  float t[12];
#pragma unroll
  for (int i = 0; i < 12; ++i) t[i] = 0.4f * (float)i - 2.2f;
  float b[11];
#pragma unroll
  for (int i = 0; i < 11; ++i)
    b[i] = (x >= t[i] && x < t[i + 1]) ? 1.0f : 0.0f;
#pragma unroll
  for (int p = 1; p <= 3; ++p) {
    const float inv = 2.5f / (float)p;   // 1 / (p * h), h = 0.4
#pragma unroll
    for (int i = 0; i < 11 - p; ++i) {
      float left  = (x - t[i]) * inv;
      float right = (t[i + p + 1] - x) * inv;
      b[i] = left * b[i] + right * b[i + 1];
    }
  }
#pragma unroll
  for (int i = 0; i < NCOEF; ++i) r[i] = b[i];
}

__device__ __forceinline__ ushort8 bases8_bf(float x) {
  float bs[NCOEF];
  bases8(x, bs);
  ushort8 p;
#pragma unroll
  for (int g = 0; g < NCOEF; ++g) p[g] = f2bf(bs[g]);
  return p;
}

// Write fused features for one scalar into a K-major bf16 row.
__device__ __forceinline__ void write_feat(unsigned short* __restrict__ row,
                                           int silu_idx, int spl_idx, float x) {
  row[silu_idx] = f2bf(silu(x));
  *(ushort8*)(row + spl_idx) = bases8_bf(x);   // 16B aligned by construction
}

// ---------------- prep: fuse + quantize weights to WMMA B layout ----------------

__global__ void __launch_bounds__(256)
prep_weights(const float* __restrict__ base_w1, const float* __restrict__ spline_w1,
             const float* __restrict__ base_w2, const float* __restrict__ spline_w2,
             unsigned short* __restrict__ wf1, unsigned short* __restrict__ wf2) {
  int idx = blockIdx.x * 256 + threadIdx.x;
  const int total1 = HID * K1;
  if (idx < total1) {
    int o = idx / K1, k = idx % K1;
    float v = 0.0f;
    if (k < 524)                       v = base_w1[o * 524 + k];
    else if (k >= SPL1_OFF && k < SPL1_OFF + 524 * NCOEF)
                                       v = spline_w1[o * (524 * NCOEF) + (k - SPL1_OFF)];
    wf1[idx] = f2bf(v);
  } else {
    int j = idx - total1;
    if (j < HID * K2) {
      int o = j / K2, k = j % K2;
      float v = (k < SPL2_OFF) ? base_w2[o * HID + k]
                               : spline_w2[o * (HID * NCOEF) + (k - SPL2_OFF)];
      wf2[j] = f2bf(v);
    }
  }
}

// ---------------- zero-fill feature buffers (covers alignment gaps/pads) -------

__global__ void __launch_bounds__(256)
zero_ws(uint4v* __restrict__ p, int n16) {
  int i = blockIdx.x * 256 + threadIdx.x;
  if (i < n16) p[i] = uint4v{0u, 0u, 0u, 0u};
}

// ---------------- feat1 for t = 0 (h0 = 0) ------------------------------------

__global__ void __launch_bounds__(256)
init_feat1(const float* __restrict__ x_seq, unsigned short* __restrict__ feat1) {
  int idx = blockIdx.x * 256 + threadIdx.x;
  if (idx >= BATCH * 524) return;
  int row = idx / 524, j = idx % 524;
  float x = (j < D_IN) ? x_seq[(size_t)row * SEQ_T * D_IN + j] : 0.0f;  // t=0
  write_feat(feat1 + (size_t)row * K1, j, SPL1_OFF + j * NCOEF, x);
}

// ---------------- GEMM wave-tile machinery -------------------------------------
// Wave tile: 16 rows x 32 cols. 1024 waves = 128 blocks x 256 threads.
// A fragment (16x32 bf16): lane = {row = lane&15, K-half = lane>>4}, 16
//   contiguous K halfs per lane -> 32B aligned load from row-major K-contiguous.
// B fragment (32x16 bf16): lane = {col = lane&15, K-half = lane>>4}, same
//   pattern from N-major K-contiguous weights.
// C/D (16x16 f32): VGPR r, lane L -> (m = (L>>4)*8 + r, n = L&15).

template <int K>
__device__ __forceinline__ void wave_gemm(const unsigned short* __restrict__ feat,
                                          const unsigned short* __restrict__ w,
                                          int m0, int n0, int lane,
                                          v8f& c0, v8f& c1) {
  const int lrow = lane & 15, khalf = lane >> 4;
  const unsigned short* ap = feat + (size_t)(m0 + lrow) * K + khalf * 16;
  const unsigned short* bp0 = w + (size_t)(n0 + lrow) * K + khalf * 16;
  const unsigned short* bp1 = bp0 + (size_t)16 * K;
  c0 = v8f{};
  c1 = v8f{};
#pragma unroll 4
  for (int k = 0; k < K; k += 32) {
    v16bf a  = *(const v16bf*)(ap + k);
    v16bf b0 = *(const v16bf*)(bp0 + k);
    v16bf b1 = *(const v16bf*)(bp1 + k);
    c0 = __builtin_amdgcn_wmma_f32_16x16x32_bf16(false, a, false, b0,
                                                 (short)0, c0, false, false);
    c1 = __builtin_amdgcn_wmma_f32_16x16x32_bf16(false, a, false, b1,
                                                 (short)0, c1, false, false);
  }
}

// ---------------- layer 1: h1 = KAN1(feat1); epilogue writes feat2 -------------

__global__ void __launch_bounds__(256)
gemm1(const unsigned short* __restrict__ feat1,
      const unsigned short* __restrict__ wf1,
      unsigned short* __restrict__ feat2) {
  const int wave = (blockIdx.x * blockDim.x + threadIdx.x) >> 5;
  const int lane = threadIdx.x & 31;
  const int m0 = (wave & 63) * 16, n0 = (wave >> 6) * 32;
  v8f c0, c1;
  wave_gemm<K1>(feat1, wf1, m0, n0, lane, c0, c1);

  const int nA = n0 + (lane & 15);
  const int nB = nA + 16;
  const int mb = m0 + (lane >> 4) * 8;
#pragma unroll
  for (int r = 0; r < 8; ++r) {
    unsigned short* fr = feat2 + (size_t)(mb + r) * K2;
    write_feat(fr, nA, SPL2_OFF + nA * NCOEF, c0[r]);
    write_feat(fr, nB, SPL2_OFF + nB * NCOEF, c1[r]);
  }
}

// ---------------- layer 2: h = tanh(KAN2(feat2)); epilogue writes feat1(t+1) ---

__global__ void __launch_bounds__(256)
gemm2(const unsigned short* __restrict__ feat2,
      const unsigned short* __restrict__ wf2,
      const float* __restrict__ x_seq,
      unsigned short* __restrict__ feat1,
      float* __restrict__ h, int t) {
  const int wave = (blockIdx.x * blockDim.x + threadIdx.x) >> 5;
  const int lane = threadIdx.x & 31;
  const int m0 = (wave & 63) * 16, n0 = (wave >> 6) * 32;
  v8f c0, c1;
  wave_gemm<K2>(feat2, wf2, m0, n0, lane, c0, c1);

  const int nA = n0 + (lane & 15);
  const int nB = nA + 16;
  const int mb = m0 + (lane >> 4) * 8;
#pragma unroll
  for (int r = 0; r < 8; ++r) {
    const int m = mb + r;
    float vA = tanhf(c0[r]);
    float vB = tanhf(c1[r]);
    h[(size_t)m * HID + nA] = vA;
    h[(size_t)m * HID + nB] = vB;
    unsigned short* fr = feat1 + (size_t)m * K1;   // z = [x_{t+1}, h]
    write_feat(fr, D_IN + nA, SPL1_OFF + (D_IN + nA) * NCOEF, vA);
    write_feat(fr, D_IN + nB, SPL1_OFF + (D_IN + nB) * NCOEF, vB);
  }

  // n-tile 0 waves also stage the x part of feat1 for step t+1.
  if ((wave >> 6) == 0 && (t + 1) < SEQ_T) {
    // 16 rows * 12 x-values = 192 elements; 6 per lane.
#pragma unroll
    for (int i = 0; i < 6; ++i) {
      int e = lane * 6 + i;
      int row = m0 + e / D_IN;
      int j = e % D_IN;
      float x = x_seq[(size_t)row * SEQ_T * D_IN + (size_t)(t + 1) * D_IN + j];
      write_feat(feat1 + (size_t)row * K1, j, SPL1_OFF + j * NCOEF, x);
    }
  }
}

// ---------------- final projection: out = h @ w_out^T + b_out ------------------

__global__ void __launch_bounds__(256)
proj(const float* __restrict__ h, const float* __restrict__ w_out,
     const float* __restrict__ b_out, float* __restrict__ out) {
  int b = blockIdx.x * 256 + threadIdx.x;
  if (b >= BATCH) return;
  float s0 = b_out[0], s1 = b_out[1], s2 = b_out[2];
  const float* hr = h + (size_t)b * HID;
#pragma unroll 4
  for (int k = 0; k < HID; ++k) {
    float hv = hr[k];
    s0 += hv * w_out[k];
    s1 += hv * w_out[HID + k];
    s2 += hv * w_out[2 * HID + k];
  }
  out[b * 3 + 0] = s0;
  out[b * 3 + 1] = s1;
  out[b * 3 + 2] = s2;
}

// ---------------- host orchestration -------------------------------------------

extern "C" void kernel_launch(void* const* d_in, const int* in_sizes, int n_in,
                              void* d_out, int out_size, void* d_ws, size_t ws_size,
                              hipStream_t stream) {
  const float* x_seq     = (const float*)d_in[0];
  const float* base_w1   = (const float*)d_in[1];
  const float* spline_w1 = (const float*)d_in[2];
  const float* base_w2   = (const float*)d_in[3];
  const float* spline_w2 = (const float*)d_in[4];
  const float* w_out     = (const float*)d_in[5];
  const float* b_out     = (const float*)d_in[6];
  float* out = (float*)d_out;

  // Workspace layout (bytes):
  const size_t wf1_b   = (size_t)HID * K1 * 2;        //  4,849,664
  const size_t wf2_b   = (size_t)HID * K2 * 2;        //  4,718,592
  const size_t feat1_b = (size_t)BATCH * K1 * 2;      //  9,699,328
  const size_t feat2_b = (size_t)BATCH * K2 * 2;      //  9,437,184
  const size_t h_b     = (size_t)BATCH * HID * 4;     //  2,097,152
  const size_t need = wf1_b + wf2_b + feat1_b + feat2_b + h_b;   // ~30.8 MB
  if (ws_size < need) return;   // cannot run without scratch

  char* ws = (char*)d_ws;
  unsigned short* wf1   = (unsigned short*)ws;
  unsigned short* wf2   = (unsigned short*)(ws + wf1_b);
  unsigned short* feat1 = (unsigned short*)(ws + wf1_b + wf2_b);
  unsigned short* feat2 = (unsigned short*)(ws + wf1_b + wf2_b + feat1_b);
  float*          hbuf  = (float*)(ws + wf1_b + wf2_b + feat1_b + feat2_b);

  // 1) fuse + quantize weights (idempotent, every call).
  {
    int total = HID * K1 + HID * K2;                 // 4,784,128
    prep_weights<<<(total + 255) / 256, 256, 0, stream>>>(
        base_w1, spline_w1, base_w2, spline_w2, wf1, wf2);
  }
  // 2) zero feat1+feat2 (contiguous) so alignment gaps/pads stay zero.
  {
    int n16 = (int)((feat1_b + feat2_b) / 16);       // 1,196,032
    zero_ws<<<(n16 + 255) / 256, 256, 0, stream>>>((uint4v*)feat1, n16);
  }
  // 3) feat1 for t=0 (h0 = 0).
  {
    int total = BATCH * 524;
    init_feat1<<<(total + 255) / 256, 256, 0, stream>>>(x_seq, feat1);
  }
  // 4) sequential recurrence: 2 WMMA GEMMs per step, stream-ordered.
  for (int t = 0; t < SEQ_T; ++t) {
    gemm1<<<128, 256, 0, stream>>>(feat1, wf1, feat2);
    gemm2<<<128, 256, 0, stream>>>(feat2, wf2, x_seq, feat1, hbuf, t);
  }
  // 5) output projection.
  proj<<<(BATCH + 255) / 256, 256, 0, stream>>>(hbuf, w_out, b_out, out);
}
